// GATModel_34600256537256
// MI455X (gfx1250) — matrix operand order repaired and checked by reading, not compiled, not run on hardware
//
#include <hip/hip_runtime.h>
#include <cstdint>
#include <cstddef>

// GAT 2-layer pipeline for gfx1250 (MI455X).
//  - GEMMs: v_wmma_f32_16x16x32_f16 (f32 I/O, f16 multiplicands), one wave per
//    16x16 tile; weight matrix staged in LDS via Tensor Data Mover
//    (tensor_load_to_lds + s_wait_tensorcnt), A rows loaded with b128.
//  - Graph softmax/aggregation: float atomics; working set is L2-resident.

typedef _Float16 v16h __attribute__((ext_vector_type(16)));
typedef float    v8f  __attribute__((ext_vector_type(8)));
typedef unsigned int uint32x4 __attribute__((ext_vector_type(4)));
typedef int          int32x8  __attribute__((ext_vector_type(8)));
typedef int          int32x4  __attribute__((ext_vector_type(4)));

#define NNODES 100000

__device__ __forceinline__ void atomic_max_f32(float* addr, float v) {
  // Monotonic int trick: works for mixed-sign floats with -inf init.
  if (v >= 0.0f) atomicMax((int*)addr, __float_as_int(v));
  else           atomicMin((unsigned int*)addr, __float_as_uint(v));
}

// ---------------------------------------------------------------------------
// TDM: DMA a 2D row-major f32 tile [dim1 rows x dim0 cols] (stride0 = row
// stride in elements) from global memory into LDS at lds_off. Issued once by
// the calling wave; caller must s_wait_tensorcnt(0) then barrier.
// Descriptor layout per CDNA5 ISA 8.3/8.4 (D# group0 + group1; 2D => groups
// 2/3 zero). This toolchain exposes the 6-arg builtin form.
// ---------------------------------------------------------------------------
__device__ __forceinline__ void tdm_load_2d_f32(const float* gptr, unsigned lds_off,
                                                unsigned dim0, unsigned dim1,
                                                unsigned stride0)
{
  const unsigned long long ga = (unsigned long long)(uintptr_t)gptr;
  uint32x4 g0;
  g0[0] = 1u;                                        // count=1 (valid user D#)
  g0[1] = lds_off;                                   // lds_addr (bytes)
  g0[2] = (unsigned)(ga & 0xffffffffu);              // global_addr[31:0]
  g0[3] = (unsigned)((ga >> 32) & 0x01ffffffu)       // global_addr[56:32]
          | (2u << 30);                              // type = 2 ("image")
  int32x8 g1;
  g1[0] = (int)(2u << 16);                           // data_size = 4 bytes
  g1[1] = (int)(dim0 << 16);                         // tensor_dim0[15:0]
  g1[2] = (int)((dim0 >> 16) | (dim1 << 16));        // dim0 hi | tensor_dim1 lo
  g1[3] = (int)((dim1 >> 16) | (dim0 << 16));        // dim1 hi | tile_dim0 (=dim0)
  g1[4] = (int)(dim1 & 0xffffu);                     // tile_dim1 (=dim1), tile_dim2=0
  g1[5] = (int)stride0;                              // tensor_dim0_stride[31:0]
  g1[6] = 0;                                         // stride hi | dim1_stride lo
  g1[7] = 0;
  int32x4 z4 = {};
  int32x8 z8 = {};
  __builtin_amdgcn_tensor_load_to_lds(g0, g1, z4, z4, z8, 0);
}

// ---------------------------------------------------------------------------
// WMMA GEMM: C[M x NC] = A[M x K] * B[K x NC] (+ optional bias), one wave per
// 16x16 output tile. NCPAD = NC rounded up to 16 (only differs for NC=8).
// B is TDM-staged in LDS once per block; A rows fetched with global b128.
// Fragment layouts per CDNA5 ISA 7.12.2 (16-bit A 16x32; B 32x16; f32 C 16x16).
// ---------------------------------------------------------------------------
template<int K, int NC, int NCPAD>
__global__ void wmma_gemm_kernel(const float* __restrict__ A,
                                 const float* __restrict__ B,
                                 float* __restrict__ C,
                                 const float* __restrict__ bias, int M)
{
  constexpr int NT = NCPAD / 16;
  __shared__ float shB[K * NC];

  // Stage the whole [K x NC] weight matrix into LDS via the Tensor Data Mover.
  if (threadIdx.x < 32) {                            // wave 0 only (uniform branch)
    const unsigned lds_off = (unsigned)(unsigned long long)(uintptr_t)(void*)shB;
    tdm_load_2d_f32(B, lds_off, NC, K, NC);
    __builtin_amdgcn_s_wait_tensorcnt(0);
  }
  __syncthreads();                                   // publish LDS to all waves

  const int wid   = blockIdx.x * (blockDim.x >> 5) + (threadIdx.x >> 5);
  const int lane  = threadIdx.x & 31;
  const int tileM = wid / NT;
  const int tileN = wid % NT;
  if (tileM * 16 >= M) return;        // wave-uniform: EXEC stays all-ones for WMMA

  const int hi = lane >> 4;           // lane half (0: lanes 0-15, 1: lanes 16-31)
  const int lr = lane & 15;
  const int m  = tileM * 16 + lr;     // A row owned by this lane
  const int n  = tileN * 16 + lr;     // B/C column owned by this lane
  const bool nvalid = (n < NC);

  const float* __restrict__ arow = A + (size_t)m * K;

  v8f acc = {};
#pragma unroll
  for (int kb = 0; kb < K; kb += 32) {
    // A 16x32 f16 fragment: lane needs k in {kb+hi*8..+7} and {kb+16+hi*8..+7}
    // -> two contiguous 8-float runs, four global_load_b128.
    const float4* __restrict__ ar0 = (const float4*)(arow + kb + hi * 8);
    const float4* __restrict__ ar1 = (const float4*)(arow + kb + 16 + hi * 8);
    const float4 a0 = ar0[0], a1 = ar0[1], a2 = ar1[0], a3 = ar1[1];
    v16h af;
    af[0]  = (_Float16)a0.x; af[1]  = (_Float16)a0.y;
    af[2]  = (_Float16)a0.z; af[3]  = (_Float16)a0.w;
    af[4]  = (_Float16)a1.x; af[5]  = (_Float16)a1.y;
    af[6]  = (_Float16)a1.z; af[7]  = (_Float16)a1.w;
    af[8]  = (_Float16)a2.x; af[9]  = (_Float16)a2.y;
    af[10] = (_Float16)a2.z; af[11] = (_Float16)a2.w;
    af[12] = (_Float16)a3.x; af[13] = (_Float16)a3.y;
    af[14] = (_Float16)a3.z; af[15] = (_Float16)a3.w;

    // B 32x16 f16 fragment from LDS: VGPR v holds k = kb + (v>>2)*16 + hi*8 + 2*(v&3) (+1), col n
    v16h bf;
#pragma unroll
    for (int v = 0; v < 8; ++v) {
      const int kr = kb + (v >> 2) * 16 + hi * 8 + 2 * (v & 3);
      bf[2 * v]     = (_Float16)(nvalid ? shB[kr * NC + n] : 0.0f);
      bf[2 * v + 1] = (_Float16)(nvalid ? shB[(kr + 1) * NC + n] : 0.0f);
    }
    acc = __builtin_amdgcn_wmma_f32_16x16x32_f16(false, af, false, bf,
                                                 (short)0, acc, false, false);
  }

  if (nvalid) {
    const float bv = bias ? bias[n] : 0.0f;
#pragma unroll
    for (int r = 0; r < 8; ++r) {
      // C layout: VGPR r -> row = tile + r (lanes 0-15) / tile + r + 8 (lanes 16-31)
      const int row = tileM * 16 + hi * 8 + r;
      C[(size_t)row * NC + n] = acc[r] + bv;
    }
  }
}

// ---------------------------------------------------------------------------
// Per (node, head): attention logits alpha_src/alpha_dst; init segment max/sum.
// ---------------------------------------------------------------------------
template<int H, int C>
__global__ void alpha_init_kernel(const float* __restrict__ h,
                                  const float* __restrict__ a_src,
                                  const float* __restrict__ a_dst,
                                  float* __restrict__ as, float* __restrict__ ad,
                                  float* __restrict__ mbuf, float* __restrict__ sbuf)
{
  const int tid = blockIdx.x * blockDim.x + threadIdx.x;
  if (tid >= NNODES * H) return;
  const int nn = tid / H, hh = tid % H;
  const float* __restrict__ hr = h + (size_t)nn * H * C + hh * C;
  float ss = 0.0f, dd = 0.0f;
#pragma unroll
  for (int c = 0; c < C; ++c) {
    const float hv = hr[c];
    ss += hv * a_src[hh * C + c];
    dd += hv * a_dst[hh * C + c];
  }
  as[tid] = ss;
  ad[tid] = dd;
  mbuf[tid] = __int_as_float(0xff800000u);  // -inf
  sbuf[tid] = 0.0f;
}

__global__ void zero_kernel(float* __restrict__ p, int n) {
  const int i = blockIdx.x * blockDim.x + threadIdx.x;
  if (i < n) p[i] = 0.0f;
}

// ---------------------------------------------------------------------------
// Edge passes. Edge t < E comes from edge_index; t in [E, E+N) is a self-loop.
// ---------------------------------------------------------------------------
template<int H>
__global__ void edge_max_kernel(const long long* __restrict__ ei, int E,
                                const float* __restrict__ as,
                                const float* __restrict__ ad,
                                float* __restrict__ mbuf)
{
  const int t = blockIdx.x * blockDim.x + threadIdx.x;
  const int ET = E + NNODES;
  if (t >= ET) return;
  int src, dst;
  if (t < E) { src = (int)ei[t]; dst = (int)ei[(size_t)E + t]; }
  else       { src = dst = t - E; }
#pragma unroll
  for (int hh = 0; hh < H; ++hh) {
    float e = as[src * H + hh] + ad[dst * H + hh];
    e = (e > 0.0f) ? e : 0.2f * e;                 // leaky_relu(0.2)
    atomic_max_f32(&mbuf[dst * H + hh], e);
  }
}

template<int H>
__global__ void edge_sum_kernel(const long long* __restrict__ ei, int E,
                                const float* __restrict__ as,
                                const float* __restrict__ ad,
                                const float* __restrict__ mbuf,
                                float* __restrict__ sbuf)
{
  const int t = blockIdx.x * blockDim.x + threadIdx.x;
  const int ET = E + NNODES;
  if (t >= ET) return;
  int src, dst;
  if (t < E) { src = (int)ei[t]; dst = (int)ei[(size_t)E + t]; }
  else       { src = dst = t - E; }
#pragma unroll
  for (int hh = 0; hh < H; ++hh) {
    float e = as[src * H + hh] + ad[dst * H + hh];
    e = (e > 0.0f) ? e : 0.2f * e;
    atomicAdd(&sbuf[dst * H + hh], __expf(e - mbuf[dst * H + hh]));
  }
}

// One wave per edge: lane l owns channel l of each head (C == 32 == wave32).
template<int H>
__global__ void edge_aggr_kernel(const long long* __restrict__ ei, int E,
                                 const float* __restrict__ as,
                                 const float* __restrict__ ad,
                                 const float* __restrict__ mbuf,
                                 const float* __restrict__ sbuf,
                                 const float* __restrict__ h,
                                 float* __restrict__ out)
{
  const int wid  = blockIdx.x * (blockDim.x >> 5) + (threadIdx.x >> 5);
  const int lane = threadIdx.x & 31;
  const int ET = E + NNODES;
  if (wid >= ET) return;
  int src, dst;
  if (wid < E) { src = (int)ei[wid]; dst = (int)ei[(size_t)E + wid]; }
  else         { src = dst = wid - E; }
  const float* __restrict__ hs = h + (size_t)src * H * 32;
  float* __restrict__ od = out + (size_t)dst * H * 32;
  __builtin_prefetch(hs + lane, 0, 0);             // global_prefetch_b8
#pragma unroll
  for (int hh = 0; hh < H; ++hh) {
    float e = as[src * H + hh] + ad[dst * H + hh];
    e = (e > 0.0f) ? e : 0.2f * e;
    const float alpha = __expf(e - mbuf[dst * H + hh]) /
                        (sbuf[dst * H + hh] + 1e-16f);
    atomicAdd(&od[hh * 32 + lane], hs[hh * 32 + lane] * alpha);
  }
}

template<int HC>
__global__ void bias_relu_kernel(float* __restrict__ x, const float* __restrict__ b)
{
  const int i = blockIdx.x * blockDim.x + threadIdx.x;
  if (i >= NNODES * HC) return;
  const float v = x[i] + b[i % HC];
  x[i] = v > 0.0f ? v : 0.0f;
}

// ---------------------------------------------------------------------------
extern "C" void kernel_launch(void* const* d_in, const int* in_sizes, int n_in,
                              void* d_out, int out_size, void* d_ws, size_t ws_size,
                              hipStream_t stream)
{
  const float* x    = (const float*)d_in[0];
  const float* W1   = (const float*)d_in[1];
  const float* aS1  = (const float*)d_in[2];
  const float* aD1  = (const float*)d_in[3];
  const float* b1   = (const float*)d_in[4];
  const float* W2   = (const float*)d_in[5];
  const float* aS2  = (const float*)d_in[6];
  const float* aD2  = (const float*)d_in[7];
  const float* b2   = (const float*)d_in[8];
  const float* Wl   = (const float*)d_in[9];
  const float* bl   = (const float*)d_in[10];
  const long long* ei = (const long long*)d_in[11];  // int64 [2, E]

  const int E  = in_sizes[11] / 2;
  const int ET = E + NNODES;

  // Workspace layout (floats). Layer-2 buffers alias the dead h1 region.
  float* w    = (float*)d_ws;
  float* h1   = w;                                   // N*128
  float* out1 = w + (size_t)NNODES * 128;            // N*128
  float* as1  = out1 + (size_t)NNODES * 128;         // N*4
  float* ad1  = as1 + (size_t)NNODES * 4;            // N*4
  float* m1   = ad1 + (size_t)NNODES * 4;            // N*4
  float* s1   = m1  + (size_t)NNODES * 4;            // N*4
  float* h2   = h1;                                  // N*32   (aliases h1)
  float* out2 = h1 + (size_t)NNODES * 32;            // N*32
  float* as2  = h1 + (size_t)NNODES * 64;            // N
  float* ad2  = as2 + NNODES;                        // N
  float* m2   = ad2 + NNODES;                        // N
  float* s2   = m2  + NNODES;                        // N

  const int TB = 256;
  const int edgeBlocks = (ET + TB - 1) / TB;
  const int aggrBlocks = (ET + 7) / 8;               // 8 waves per block

  // ---- Layer 1 (H=4, C=32) ----
  // h1 = x @ W1 : [100000x64]@[64x128]  -> 6250*8 tiles, 8 waves/block
  wmma_gemm_kernel<64, 128, 128><<<6250, TB, 0, stream>>>(x, W1, h1, nullptr, NNODES);
  alpha_init_kernel<4, 32><<<(NNODES * 4 + TB - 1) / TB, TB, 0, stream>>>(
      h1, aS1, aD1, as1, ad1, m1, s1);
  zero_kernel<<<(NNODES * 128 + TB - 1) / TB, TB, 0, stream>>>(out1, NNODES * 128);
  edge_max_kernel<4><<<edgeBlocks, TB, 0, stream>>>(ei, E, as1, ad1, m1);
  edge_sum_kernel<4><<<edgeBlocks, TB, 0, stream>>>(ei, E, as1, ad1, m1, s1);
  edge_aggr_kernel<4><<<aggrBlocks, TB, 0, stream>>>(ei, E, as1, ad1, m1, s1, h1, out1);
  bias_relu_kernel<128><<<(NNODES * 128 + TB - 1) / TB, TB, 0, stream>>>(out1, b1);

  // ---- Layer 2 (H=1, C=32) ----
  // h2 = out1 @ W2 : [100000x128]@[128x32] -> 6250*2 tiles
  wmma_gemm_kernel<128, 32, 32><<<(6250 * 2 + 7) / 8, TB, 0, stream>>>(
      out1, W2, h2, nullptr, NNODES);
  alpha_init_kernel<1, 32><<<(NNODES + TB - 1) / TB, TB, 0, stream>>>(
      h2, aS2, aD2, as2, ad2, m2, s2);
  zero_kernel<<<(NNODES * 32 + TB - 1) / TB, TB, 0, stream>>>(out2, NNODES * 32);
  edge_max_kernel<1><<<edgeBlocks, TB, 0, stream>>>(ei, E, as2, ad2, m2);
  edge_sum_kernel<1><<<edgeBlocks, TB, 0, stream>>>(ei, E, as2, ad2, m2, s2);
  edge_aggr_kernel<1><<<aggrBlocks, TB, 0, stream>>>(ei, E, as2, ad2, m2, s2, h2, out2);
  bias_relu_kernel<32><<<(NNODES * 32 + TB - 1) / TB, TB, 0, stream>>>(out2, b2);

  // ---- Final linear: d_out = out2 @ Wl + bl : [100000x32]@[32x8] (pad N to 16) ----
  wmma_gemm_kernel<32, 8, 16><<<(6250 + 7) / 8, TB, 0, stream>>>(
      out2, Wl, (float*)d_out, bl, NNODES);
}
